// Attention_72988674228805
// MI455X (gfx1250) — matrix operand order repaired and checked by reading, compiled, not run on hardware
//
#include <hip/hip_runtime.h>

// ---------------------------------------------------------------------------
// Problem constants (from reference): b=4, n=1024, dim=512, h=8, d=64
// ---------------------------------------------------------------------------
#define B_     4
#define N_     1024
#define DIM_   512
#define H_     8
#define D_     64
#define INNER_ 512           // h*d
#define QKV_N  1536          // 3*inner
#define SCALE_ 0.125f        // 64^-0.5

typedef __bf16 bf16;
typedef __attribute__((ext_vector_type(16))) __bf16        v16bf;
typedef __attribute__((ext_vector_type(2)))  __bf16        v2bf;
typedef __attribute__((ext_vector_type(8)))  float         v8f;
typedef __attribute__((ext_vector_type(4)))  float         v4f;
typedef __attribute__((ext_vector_type(2)))  float         v2f;
typedef __attribute__((ext_vector_type(4)))  unsigned int  v4u;
typedef __attribute__((ext_vector_type(8)))  unsigned int  v8u;

static __device__ __forceinline__ v8f wmma_bf16(v16bf a, v16bf b, v8f c) {
  // D = A(16x32 bf16) x B(32x16 bf16) + C(16x16 f32)
  return __builtin_amdgcn_wmma_f32_16x16x32_bf16(
      /*neg_a=*/false, a, /*neg_b=*/false, b,
      /*c_mod=*/(short)0, c, /*reuse_a=*/false, /*reuse_b=*/false);
}

// ---------------------------------------------------------------------------
// Fragment loaders (wave32) using contiguous k order: lane L, element s -> k = 16*(L>>4) + s.
// A (16x32, 16-bit): lane L -> row = L&15 ; K-halves split across lane pairs;
//   VGPRs 0-3 = 16 contiguous bytes at K=kb, VGPRs 4-7 = 16 bytes at K=kb+8 (kb = 0 or 16)
//   -> two 128-bit loads per lane.
// B (32x16, 16-bit): lane L -> col = L&15 ; lanes 0-15 K 0-15, lanes 16-31
//   K 16-31, pairs per VGPR.
// ---------------------------------------------------------------------------
static __device__ __forceinline__ v16bf load_a_bf16(const bf16* base, int ld, int lane) {
  const int row = lane & 15;
  const int kb  = (lane & 16) ? 16 : 0;
  const bf16* p = base + (size_t)row * ld + kb;
  v4u lo = *(const v4u*)p;            // K kb .. kb+7
  v4u hi = *(const v4u*)(p + 8);      // K kb+8 .. kb+15
  v8u u;
  u[0] = lo[0]; u[1] = lo[1]; u[2] = lo[2]; u[3] = lo[3];
  u[4] = hi[0]; u[5] = hi[1]; u[6] = hi[2]; u[7] = hi[3];
  return __builtin_bit_cast(v16bf, u);
}

static __device__ __forceinline__ v16bf load_a_f32(const float* base, int ld, int lane) {
  const int row = lane & 15;
  const int kb  = (lane & 16) ? 16 : 0;
  const float* p = base + (size_t)row * ld + kb;
  v4f f0 = *(const v4f*)p;            // K kb..kb+3
  v4f f1 = *(const v4f*)(p + 4);      // K kb+4..kb+7
  v4f f2 = *(const v4f*)(p + 8);      // K kb+8..
  v4f f3 = *(const v4f*)(p + 12);
  v16bf a;
#pragma unroll
  for (int i = 0; i < 4; ++i) {
    a[i]      = (bf16)f0[i];
    a[4 + i]  = (bf16)f1[i];
    a[8 + i]  = (bf16)f2[i];
    a[12 + i] = (bf16)f3[i];
  }
  return a;
}

// B from row-major K x N source (ld = row stride), f32 -> bf16 (strided cols)
static __device__ __forceinline__ v16bf load_b_f32(const float* base, int ld, int lane) {
  const int col = lane & 15;
  const int kb  = (lane & 16) ? 16 : 0;
  v16bf b;
#pragma unroll
  for (int r = 0; r < 8; ++r) {
    const int k = kb + 2 * r;
    b[2 * r]     = (bf16)base[(size_t)k * ld + col];
    b[2 * r + 1] = (bf16)base[(size_t)(k + 1) * ld + col];
  }
  return b;
}

// B from row-major K x N source, bf16 (strided cols)
static __device__ __forceinline__ v16bf load_b_bf16(const bf16* base, int ld, int lane) {
  const int col = lane & 15;
  const int kb  = (lane & 16) ? 16 : 0;
  v16bf b;
#pragma unroll
  for (int r = 0; r < 8; ++r) {
    const int k = kb + 2 * r;
    b[2 * r]     = base[(size_t)k * ld + col];
    b[2 * r + 1] = base[(size_t)(k + 1) * ld + col];
  }
  return b;
}

// B from row-major N x K source (B^T in memory): element(k,col)=base[col*ld+k]
// Each lane needs 16 contiguous K elements -> two 128-bit loads.
static __device__ __forceinline__ v16bf load_b_nt_bf16(const bf16* base, int ld, int lane) {
  const int col = lane & 15;
  const int kb  = (lane & 16) ? 16 : 0;
  const bf16* p = base + (size_t)col * ld + kb;
  v4u lo = *(const v4u*)p;            // K kb..kb+7
  v4u hi = *(const v4u*)(p + 8);      // K kb+8..kb+15
  v8u u;
  u[0] = lo[0]; u[1] = lo[1]; u[2] = lo[2]; u[3] = lo[3];
  u[4] = hi[0]; u[5] = hi[1]; u[6] = hi[2]; u[7] = hi[3];
  return __builtin_bit_cast(v16bf, u);
}

static __device__ __forceinline__ float wave_max(float x) {
#pragma unroll
  for (int off = 16; off; off >>= 1) x = fmaxf(x, __shfl_xor(x, off, 32));
  return x;
}
static __device__ __forceinline__ float wave_sum(float x) {
#pragma unroll
  for (int off = 16; off; off >>= 1) x += __shfl_xor(x, off, 32);
  return x;
}

// ---------------------------------------------------------------------------
// Kernel 1: QKV = X[4096,512] @ Wqkv[512,1536]; scatter to head-major bf16
// q/k/v [b,h,n,d]; q pre-scaled by SCALE.  Block tile 64x64 (8 waves, each
// wave 16x32).  grid = (1536/64, 4096/64); blockIdx.x>>3 selects q/k/v
// (block-uniform -> scalar branch, no EXEC-mask scatter).
// ---------------------------------------------------------------------------
__global__ __launch_bounds__(256) void qkv_kernel(
    const float* __restrict__ x, const float* __restrict__ wq,
    bf16* __restrict__ q, bf16* __restrict__ k, bf16* __restrict__ v) {
  const int tid = threadIdx.x, w = tid >> 5, lane = tid & 31;
  const int m0 = blockIdx.y * 64 + (w & 3) * 16;
  const int n0 = blockIdx.x * 64 + (w >> 2) * 32;       // global qkv column
  v8f c0 = {}, c1 = {};
  for (int kk = 0; kk < DIM_; kk += 32) {
    __builtin_prefetch(x + (size_t)m0 * DIM_ + kk + 64, 0, 0);
    v16bf a  = load_a_f32(x + (size_t)m0 * DIM_ + kk, DIM_, lane);
    v16bf b0 = load_b_f32(wq + (size_t)kk * QKV_N + n0,      QKV_N, lane);
    v16bf b1 = load_b_f32(wq + (size_t)kk * QKV_N + n0 + 16, QKV_N, lane);
    c0 = wmma_bf16(a, b0, c0);
    c1 = wmma_bf16(a, b1, c1);
  }
  // block-uniform destination select
  const int which = blockIdx.x >> 3;                    // 0=q, 1=k, 2=v
  bf16* dst = (which == 0) ? q : (which == 1) ? k : v;
  const float scl = (which == 0) ? SCALE_ : 1.0f;
  const int inner0 = (blockIdx.x & 7) * 64 + (w >> 2) * 32;  // col within 512
  const int hi8 = (lane >> 4) << 3, col = lane & 15;
#pragma unroll
  for (int r = 0; r < 8; ++r) {
    const int row = m0 + r + hi8;
    const int bg = row >> 10, nn = row & (N_ - 1);
#pragma unroll
    for (int t = 0; t < 2; ++t) {
      const int inner = inner0 + t * 16 + col;
      const int head = inner >> 6, dd = inner & (D_ - 1);
      const float val = (t ? c1[r] : c0[r]) * scl;
      dst[(((size_t)(bg * H_ + head)) * N_ + nn) * D_ + dd] = (bf16)val;
    }
  }
}

// ---------------------------------------------------------------------------
// Kernel 2: fused  S = qK^T  ->  entmax1.5(S)  ->  out = P @ V
// One workgroup = one (b,h) and 16 query rows.  grid = (1024/16, b*h)
// LDS: 16 x 1024 f32 scores (exactly 64KB).  After entmax, bf16
// probabilities are written in-place into the low half of each row; the
// upper half doubles as reduction scratch for the K-split P@V GEMM.
// ---------------------------------------------------------------------------
__global__ __launch_bounds__(256) void attn_kernel(
    const bf16* __restrict__ qb, const bf16* __restrict__ kb,
    const bf16* __restrict__ vb, bf16* __restrict__ ao) {
  __shared__ float S[16][N_];                      // 64 KB

  const int tid = threadIdx.x, w = tid >> 5, lane = tid & 31;
  const int bh = blockIdx.y;
  const int i0 = blockIdx.x * 16;
  const bf16* qh = qb + (size_t)bh * N_ * D_;
  const bf16* kh = kb + (size_t)bh * N_ * D_;
  const bf16* vh = vb + (size_t)bh * N_ * D_;
  const int hi8 = (lane >> 4) << 3, col = lane & 15;

  // ---- Phase A: S[16,1024] = q_tile(16x64) @ K^T --------------------------
  {
    v16bf a0 = load_a_bf16(qh + (size_t)i0 * D_,      D_, lane);
    v16bf a1 = load_a_bf16(qh + (size_t)i0 * D_ + 32, D_, lane);
#pragma unroll
    for (int t = 0; t < 8; ++t) {
      const int n0 = (w + 8 * t) * 16;              // key tile start
      v16bf b0 = load_b_nt_bf16(kh + (size_t)n0 * D_,      D_, lane);
      v16bf b1 = load_b_nt_bf16(kh + (size_t)n0 * D_ + 32, D_, lane);
      v8f c = {};
      c = wmma_bf16(a0, b0, c);
      c = wmma_bf16(a1, b1, c);
#pragma unroll
      for (int r = 0; r < 8; ++r) S[r + hi8][n0 + col] = c[r];
    }
  }
  __syncthreads();

  // ---- Phase B: exact 1.5-entmax per row (bisection for tau) --------------
  // p = max(z/2 - max(z/2) - tau, 0)^2 with tau s.t. sum(p)=1; tau in [-1,0].
  // Each lane owns column pairs 2*lane + 64*t  -> b64 loads / packed stores.
#pragma unroll
  for (int rr = 0; rr < 2; ++rr) {
    const int row = w * 2 + rr;
    float vals[32];
    float m = -3.402823466e38f;
#pragma unroll
    for (int t = 0; t < 16; ++t) {
      const v2f pv = *(const v2f*)&S[row][2 * lane + 64 * t];
      vals[2 * t]     = pv[0];
      vals[2 * t + 1] = pv[1];
      m = fmaxf(m, fmaxf(pv[0], pv[1]));
    }
    m = wave_max(m);
#pragma unroll
    for (int t = 0; t < 32; ++t) vals[t] = 0.5f * (vals[t] - m);   // max -> 0
    float lo = -1.0f, hi = 0.0f;
    for (int it = 0; it < 26; ++it) {
      const float mid = 0.5f * (lo + hi);
      float s = 0.0f;
#pragma unroll
      for (int t = 0; t < 32; ++t) {
        const float d = fmaxf(vals[t] - mid, 0.0f);
        s += d * d;
      }
      s = wave_sum(s);
      if (s >= 1.0f) lo = mid; else hi = mid;
    }
    const float tau = 0.5f * (lo + hi);
    // write bf16 probabilities in-place (low 2KB of this row's 4KB)
    bf16* Prow = ((bf16*)&S[0][0]) + (size_t)row * 2048;
#pragma unroll
    for (int t = 0; t < 16; ++t) {
      const float d0 = fmaxf(vals[2 * t]     - tau, 0.0f);
      const float d1 = fmaxf(vals[2 * t + 1] - tau, 0.0f);
      v2bf pk;
      pk[0] = (bf16)(d0 * d0);
      pk[1] = (bf16)(d1 * d1);
      *(v2bf*)(Prow + 2 * lane + 64 * t) = pk;      // ds_store_b32
    }
  }
  __syncthreads();

  // ---- Phase C: out(16x64) = P(16x1024) @ V(1024x64), 2-way K split -------
  {
    const bf16* P = (const bf16*)&S[0][0];          // row stride = 2048 bf16
    const int jt = w & 3, ksplit = w >> 2;
    const int n0 = jt * 16;
    v8f c = {};
    const int kbeg = ksplit * 512;
#pragma unroll 4
    for (int kk = kbeg; kk < kbeg + 512; kk += 32) {
      v16bf a = load_a_bf16(P + kk, 2048, lane);    // ds_load_b128 x2
      v16bf b = load_b_bf16(vh + (size_t)kk * D_ + n0, D_, lane);
      c = wmma_bf16(a, b, c);
    }
    if (ksplit == 1) {
#pragma unroll
      for (int r = 0; r < 8; ++r) S[r + hi8][512 + n0 + col] = c[r];  // scratch
    }
    __syncthreads();
    if (ksplit == 0) {
      const int bg = bh >> 3, head = bh & (H_ - 1);
#pragma unroll
      for (int r = 0; r < 8; ++r) {
        const float val = c[r] + S[r + hi8][512 + n0 + col];
        const int irow = i0 + r + hi8;
        ao[((size_t)(bg * N_ + irow)) * INNER_ + head * D_ + n0 + col] = (bf16)val;
      }
    }
  }
}

// ---------------------------------------------------------------------------
// Kernel 3: out[4096,512] = AO[4096,512](bf16) @ Wout[512,512](f32) -> f32
// grid = (512/64, 4096/64)
// ---------------------------------------------------------------------------
__global__ __launch_bounds__(256) void out_kernel(
    const bf16* __restrict__ ao, const float* __restrict__ wo,
    float* __restrict__ out) {
  const int tid = threadIdx.x, w = tid >> 5, lane = tid & 31;
  const int m0 = blockIdx.y * 64 + (w & 3) * 16;
  const int n0 = blockIdx.x * 64 + (w >> 2) * 32;
  v8f c0 = {}, c1 = {};
  for (int kk = 0; kk < INNER_; kk += 32) {
    v16bf a  = load_a_bf16(ao + (size_t)m0 * INNER_ + kk, INNER_, lane);
    v16bf b0 = load_b_f32(wo + (size_t)kk * DIM_ + n0,      DIM_, lane);
    v16bf b1 = load_b_f32(wo + (size_t)kk * DIM_ + n0 + 16, DIM_, lane);
    c0 = wmma_bf16(a, b0, c0);
    c1 = wmma_bf16(a, b1, c1);
  }
  const int hi8 = (lane >> 4) << 3, col = lane & 15;
#pragma unroll
  for (int r = 0; r < 8; ++r) {
    const int row = m0 + r + hi8;
    out[(size_t)row * DIM_ + n0 + col]      = c0[r];
    out[(size_t)row * DIM_ + n0 + 16 + col] = c1[r];
  }
}

// ---------------------------------------------------------------------------
extern "C" void kernel_launch(void* const* d_in, const int* in_sizes, int n_in,
                              void* d_out, int out_size, void* d_ws, size_t ws_size,
                              hipStream_t stream) {
  const float* x  = (const float*)d_in[0];   // [4,1024,512]
  const float* wq = (const float*)d_in[1];   // [512,1536]
  const float* wo = (const float*)d_in[2];   // [512,512]
  float* out = (float*)d_out;                // [4,1024,512]

  const size_t hd = (size_t)B_ * H_ * N_ * D_;   // 2,097,152 elements
  bf16* qb = (bf16*)d_ws;                        // 4 MB
  bf16* kb = qb + hd;                            // 4 MB
  bf16* vb = kb + hd;                            // 4 MB
  bf16* ao = vb + hd;                            // 4 MB  (total 16 MB of ws)

  qkv_kernel<<<dim3(QKV_N / 64, (B_ * N_) / 64), 256, 0, stream>>>(x, wq, qb, kb, vb);
  attn_kernel<<<dim3(N_ / 16, B_ * H_), 256, 0, stream>>>(qb, kb, vb, ao);
  out_kernel<<<dim3(DIM_ / 64, (B_ * N_) / 64), 256, 0, stream>>>(ao, wo, out);
}
